// LSTMModel_32959579029778
// MI455X (gfx1250) — matrix operand order repaired and checked by reading, not compiled, not run on hardware
//
#include <hip/hip_runtime.h>

// ---------------------------------------------------------------------------
// LSTM (L=2, B=32, S=2048, H=512) for gfx1250.
// Phase 1: xg = A @ Wih^T + bias  -- LDS-tiled WMMA GEMM (128x128x32 blocks),
//          tiles staged with GLOBAL_LOAD_ASYNC_TO_LDS_B128, double-buffered.
// Phase 2: persistent recurrence, Whh slice resident in LDS, async h staging,
//          global_prefetch for next timestep's xg, spin barrier per step.
// ---------------------------------------------------------------------------

typedef __bf16 bf16;
typedef __attribute__((ext_vector_type(8)))  __bf16 v8bf;
typedef __attribute__((ext_vector_type(16))) __bf16 v16bf;
typedef __attribute__((ext_vector_type(8)))  float  v8f;

static constexpr int L  = 2;
static constexpr int B  = 32;
static constexpr int S  = 2048;
static constexpr int H  = 512;
static constexpr int G  = 4 * H;     // 2048 gate columns
static constexpr int NWG = 32;       // persistent workgroups (16 hidden units each)

// GEMM tiling
static constexpr int BM = 128, BN = 128, BK = 32;
static constexpr int LDT = 48;       // padded LDS row stride (elements); 96B rows

// dynamic LDS carve for the recurrent kernel
static constexpr int LDSW_BYTES = 4 * 16 * H * sizeof(bf16);   // 64 KB  Whh slice
static constexpr int LDSH_BYTES = B * H * sizeof(bf16);        // 32 KB  h staging
static constexpr int GBUF_BYTES = 4 * B * 17 * sizeof(float);  // ~8.7 KB gates
static constexpr int SMEM_BYTES = LDSW_BYTES + LDSH_BYTES + GBUF_BYTES;

__device__ inline v16bf cat16(v8bf lo, v8bf hi) {
  return __builtin_shufflevector(lo, hi, 0, 1, 2, 3, 4, 5, 6, 7,
                                 8, 9, 10, 11, 12, 13, 14, 15);
}

// Async copy: 16 bytes global -> LDS, tracked by ASYNCcnt (CDNA5 path).
// lds_off is the LDS byte offset (low 32 bits of a shared-memory pointer).
__device__ inline void async_copy_b128(unsigned lds_off, const void* gptr) {
  asm volatile("global_load_async_to_lds_b128 %0, %1, off"
               :: "v"(lds_off), "v"(gptr) : "memory");
}
#define WAIT_ASYNCCNT(n) asm volatile("s_wait_asynccnt " #n ::: "memory")

__device__ inline unsigned lds_off_of(const void* p) {
  return (unsigned)(size_t)p;  // shared aperture: addr[31:0] == LDS offset
}

__device__ inline float sigf(float x) {
  x = fminf(fmaxf(x, -30.0f), 30.0f);
  return 1.0f / (1.0f + __expf(-x));
}
__device__ inline float tanh_fast(float x) {
  x = fminf(fmaxf(x, -15.0f), 15.0f);
  float e = __expf(2.0f * x);
  return (e - 1.0f) / (e + 1.0f);
}

// ---------------------------------------------------------------------------
// Elementwise helpers
// ---------------------------------------------------------------------------
__global__ void f32_to_bf16_kernel(const float* __restrict__ in,
                                   bf16* __restrict__ out, size_t n) {
  size_t i = (size_t)blockIdx.x * blockDim.x + threadIdx.x;
  size_t stride = (size_t)gridDim.x * blockDim.x;
  for (; i < n; i += stride) out[i] = (bf16)in[i];
}

__global__ void bias_combine_kernel(const float* __restrict__ bi,
                                    const float* __restrict__ bh,
                                    float* __restrict__ out, int n) {
  int i = blockIdx.x * blockDim.x + threadIdx.x;
  if (i < n) out[i] = bi[i] + bh[i];
}

__global__ void init_state_kernel(bf16* __restrict__ hbuf,
                                  unsigned* __restrict__ counter, int n) {
  int i = blockIdx.x * blockDim.x + threadIdx.x;
  if (i < n) hbuf[i] = (bf16)0.0f;
  if (i == 0) *counter = 0u;
}

// ---------------------------------------------------------------------------
// Phase 1: xg[r, j] = sum_k A[r,k] * Wih[j,k] + bias[j]
//   rows r = s*B + b (65536), cols j (2048), K = 512
//   mode 0: A = Xbf with X layout [B,S,H]  -> row r reads X[b][s][:]
//   mode 1: A = seq0 (layer-0 h sequence), row-major [S*B, H]
// Block = 256 threads = 8 waves, 128x128 output tile, BK=32 chunks,
// double-buffered LDS fed by async-to-LDS copies.
// ---------------------------------------------------------------------------
__global__ __launch_bounds__(256)
void gemm_xg_kernel(const bf16* __restrict__ A, const bf16* __restrict__ W,
                    const float* __restrict__ bias, float* __restrict__ xg,
                    int mode) {
  __shared__ bf16 As[2][BM * LDT];
  __shared__ bf16 Bs[2][BN * LDT];

  const int tid  = threadIdx.x;
  const int lane = tid & 31;
  const int wave = tid >> 5;
  const int wr = wave >> 2;            // 0..1  (row half: 64 rows)
  const int wc = wave & 3;             // 0..3  (col quarter: 32 cols)
  const int r0 = blockIdx.y * BM;      // block row base
  const int j0 = blockIdx.x * BN;      // block col base

  // ---- per-thread copy assignment: 2 A rows + 2 B cols, one b128 each ----
  const int crow = tid >> 2;           // 0..63
  const int ck   = (tid & 3) * 8;      // k element offset of this thread's b128
  auto a_rowptr = [&](int rg) -> const bf16* {
    if (mode == 0) {
      int b = rg & (B - 1), s = rg >> 5;
      return A + (size_t)b * S * H + (size_t)s * H;
    }
    return A + (size_t)rg * H;
  };
  const bf16* arow0 = a_rowptr(r0 + crow);
  const bf16* arow1 = a_rowptr(r0 + crow + 64);
  const bf16* brow0 = W + (size_t)(j0 + crow) * H;
  const bf16* brow1 = W + (size_t)(j0 + crow + 64) * H;

  auto issue_chunk = [&](int kt, int buf) {
    int k0 = kt * BK;
    async_copy_b128(lds_off_of(&As[buf][crow * LDT + ck]),        arow0 + k0 + ck);
    async_copy_b128(lds_off_of(&As[buf][(crow + 64) * LDT + ck]), arow1 + k0 + ck);
    async_copy_b128(lds_off_of(&Bs[buf][crow * LDT + ck]),        brow0 + k0 + ck);
    async_copy_b128(lds_off_of(&Bs[buf][(crow + 64) * LDT + ck]), brow1 + k0 + ck);
  };

  // ---- accumulators: 4 row tiles x 2 col tiles per wave ----
  v8f acc[4][2];
#pragma unroll
  for (int m = 0; m < 4; ++m)
#pragma unroll
    for (int n = 0; n < 2; ++n) acc[m][n] = (v8f)(0.0f);

  const int akoff = (lane < 16) ? 0 : 8;   // ISA 16-bit A lane layout
  const int bkoff = (lane < 16) ? 0 : 8;   // ISA 16-bit B lane layout

  issue_chunk(0, 0);
#pragma unroll 1
  for (int kt = 0; kt < H / BK; ++kt) {
    const int buf = kt & 1;
    if (kt + 1 < H / BK) {
      issue_chunk(kt + 1, buf ^ 1);
      WAIT_ASYNCCNT(4);                 // chunk kt's 4 copies have landed
    } else {
      WAIT_ASYNCCNT(0);
    }
    __syncthreads();

#pragma unroll
    for (int m = 0; m < 4; ++m) {
      const bf16* ap = &As[buf][(wr * 64 + m * 16 + (lane & 15)) * LDT + akoff];
      v8bf a0 = *(const v8bf*)(ap);
      v8bf a1 = *(const v8bf*)(ap + 16);
      v16bf av = cat16(a0, a1);
#pragma unroll
      for (int n = 0; n < 2; ++n) {
        const bf16* bp = &Bs[buf][(wc * 32 + n * 16 + (lane & 15)) * LDT + bkoff];
        v8bf b0 = *(const v8bf*)(bp);
        v8bf b1 = *(const v8bf*)(bp + 16);
        acc[m][n] = __builtin_amdgcn_wmma_f32_16x16x32_bf16(
            false, av, false, cat16(b0, b1), (short)0, acc[m][n], false, false);
      }
    }
    __syncthreads();
  }

  // ---- epilogue: add bias, store f32 ----
#pragma unroll
  for (int n = 0; n < 2; ++n) {
    const int j = j0 + wc * 32 + n * 16 + (lane & 15);
    const float bv = bias[j];
#pragma unroll
    for (int m = 0; m < 4; ++m) {
#pragma unroll
      for (int v = 0; v < 8; ++v) {
        int r = r0 + wr * 64 + m * 16 + ((lane < 16) ? v : v + 8);
        xg[(size_t)r * G + j] = acc[m][n][v] + bv;
      }
    }
  }
}

// ---------------------------------------------------------------------------
// Phase 2: persistent recurrence. 32 WGs x 256 threads.
// WG `wg` owns hidden units [wg*16, wg*16+16): 4 gate chunks of 16 cols each.
// Whh slice (4*16 rows x 512 K, bf16) lives in LDS for all 2048 steps.
// ---------------------------------------------------------------------------
__global__ __launch_bounds__(256)
void lstm_recurrent_kernel(const float* __restrict__ xg,
                           const bf16* __restrict__ Whh,
                           bf16* __restrict__ hbuf,       // [2][B*H] double buffer
                           bf16* __restrict__ seq_bf,     // layer 0: [S][B][H] or null
                           float* __restrict__ out,       // layer 1: d_out or null
                           unsigned* __restrict__ counter) {
  extern __shared__ char smem[];
  bf16* ldsW = (bf16*)smem;                               // [4*16][512]
  bf16* ldsH = (bf16*)(smem + LDSW_BYTES);                // [32][512]
  float* gbuf = (float*)(smem + LDSW_BYTES + LDSH_BYTES); // [4][32][17]

  const int tid = threadIdx.x;
  const int lane = tid & 31;
  const int wave = tid >> 5;
  const int wg = blockIdx.x;
  const int n0 = wg * 16;                                 // hidden slice base

  // ---- load Whh slice into LDS (resident for whole sequence) ----
  for (int vi = tid; vi < (4 * 16 * H) / 8; vi += 256) {
    int row_local = vi >> 6;
    int k = (vi & 63) << 3;
    int g = row_local >> 4, nl = row_local & 15;
    *(v8bf*)(ldsW + row_local * H + k) =
        *(const v8bf*)(Whh + (size_t)(g * H + n0 + nl) * H + k);
  }

  const int mt = wave & 1;                                // batch row tile
  const int g = wave >> 1;                                // gate index
  const int akoff = (lane < 16) ? 0 : 8;
  const bf16* arow = ldsH + (mt * 16 + (lane & 15)) * H + akoff;
  const bf16* brow = ldsW + (g * 16 + (lane & 15)) * H + ((lane < 16) ? 0 : 8);

  float creg[2] = {0.0f, 0.0f};

  float* hT = out ? out + (size_t)B * S * H : nullptr;
  float* cT = out ? out + (size_t)B * S * H + (size_t)B * H : nullptr;

  for (int s = 0; s < S; ++s) {
    // ---- stage h_{s} into LDS via async copies ----
    const bf16* hsrc = hbuf + (size_t)(s & 1) * B * H;
    __syncthreads();
#pragma unroll
    for (int it = 0; it < (B * H) / (8 * 256); ++it) {
      int vi = tid + it * 256;
      async_copy_b128(lds_off_of(ldsH + vi * 8), hsrc + vi * 8);
    }
    WAIT_ASYNCCNT(0);
    __syncthreads();

    // ---- gates tile: acc = xg[s] slice, then += h @ Whh^T ----
    v8f acc;
    const float* xrow = xg + (size_t)s * B * G;
    const int jcol = g * H + n0 + (lane & 15);
#pragma unroll
    for (int v = 0; v < 8; ++v) {
      int b = mt * 16 + ((lane < 16) ? v : v + 8);
      acc[v] = xrow[(size_t)b * G + jcol];
    }
    if (s + 1 < S) {  // prefetch next step's xg slice (global_prefetch_b8)
#pragma unroll
      for (int v = 0; v < 8; ++v) {
        int b = mt * 16 + ((lane < 16) ? v : v + 8);
        __builtin_prefetch(xrow + (size_t)B * G + (size_t)b * G + jcol, 0, 1);
      }
    }
#pragma unroll
    for (int kt = 0; kt < H / 32; ++kt) {
      v8bf a0 = *(const v8bf*)(arow + kt * 32);
      v8bf a1 = *(const v8bf*)(arow + kt * 32 + 16);
      v8bf b0 = *(const v8bf*)(brow + kt * 32);
      v8bf b1 = *(const v8bf*)(brow + kt * 32 + 16);
      acc = __builtin_amdgcn_wmma_f32_16x16x32_bf16(
          false, cat16(a0, a1), false, cat16(b0, b1), (short)0, acc, false, false);
    }
#pragma unroll
    for (int v = 0; v < 8; ++v) {
      int b = mt * 16 + ((lane < 16) ? v : v + 8);
      gbuf[(g * 32 + b) * 17 + (lane & 15)] = acc[v];
    }
    __syncthreads();

    // ---- elementwise LSTM cell ----
    bf16* hnext = hbuf + (size_t)((s + 1) & 1) * B * H;
#pragma unroll
    for (int i = 0; i < 2; ++i) {
      int e = tid + i * 256;
      int b = e >> 4;
      int nl = e & 15;
      float ig = gbuf[(0 * 32 + b) * 17 + nl];
      float fg = gbuf[(1 * 32 + b) * 17 + nl];
      float gg = gbuf[(2 * 32 + b) * 17 + nl];
      float og = gbuf[(3 * 32 + b) * 17 + nl];
      float c = sigf(fg) * creg[i] + sigf(ig) * tanh_fast(gg);
      creg[i] = c;
      float h = sigf(og) * tanh_fast(c);
      int n = n0 + nl;
      hnext[b * H + n] = (bf16)h;
      if (seq_bf) seq_bf[((size_t)s * B + b) * H + n] = (bf16)h;
      if (out) {
        out[(size_t)b * S * H + (size_t)s * H + n] = h;
        if (s == S - 1) {
          hT[b * H + n] = h;
          cT[b * H + n] = c;
        }
      }
    }

    // ---- device-wide barrier before next step reads hnext ----
    __syncthreads();
    if (tid == 0) {
      __threadfence();
      atomicAdd(counter, 1u);
      unsigned target = (unsigned)(s + 1) * (unsigned)NWG;
      while (__hip_atomic_load(counter, __ATOMIC_RELAXED,
                               __HIP_MEMORY_SCOPE_AGENT) < target) {
      }
      __threadfence();
    }
    __syncthreads();
  }
}

// ---------------------------------------------------------------------------
// Host orchestration
// ---------------------------------------------------------------------------
extern "C" void kernel_launch(void* const* d_in, const int* in_sizes, int n_in,
                              void* d_out, int out_size, void* d_ws,
                              size_t ws_size, hipStream_t stream) {
  (void)in_sizes; (void)n_in; (void)out_size; (void)ws_size;
  const float* X   = (const float*)d_in[0];
  const float* Wih = (const float*)d_in[1];
  const float* Whh = (const float*)d_in[2];
  const float* bih = (const float*)d_in[3];
  const float* bhh = (const float*)d_in[4];
  float* out = (float*)d_out;

  char* p = (char*)d_ws;
  auto carve = [&](size_t bytes) {
    char* r = p;
    p += (bytes + 255) & ~(size_t)255;
    return r;
  };
  bf16* Xbf       = (bf16*)carve((size_t)B * S * H * sizeof(bf16));
  bf16* seq0      = (bf16*)carve((size_t)S * B * H * sizeof(bf16));
  bf16* Wihb      = (bf16*)carve((size_t)L * G * H * sizeof(bf16));
  bf16* Whhb      = (bf16*)carve((size_t)L * G * H * sizeof(bf16));
  float* bias     = (float*)carve((size_t)L * G * sizeof(float));
  float* xg       = (float*)carve((size_t)S * B * G * sizeof(float));
  bf16* hbuf      = (bf16*)carve((size_t)2 * B * H * sizeof(bf16));
  unsigned* cnt   = (unsigned*)carve(256);

  f32_to_bf16_kernel<<<4096, 256, 0, stream>>>(X, Xbf, (size_t)B * S * H);
  f32_to_bf16_kernel<<<2048, 256, 0, stream>>>(Wih, Wihb, (size_t)L * G * H);
  f32_to_bf16_kernel<<<2048, 256, 0, stream>>>(Whh, Whhb, (size_t)L * G * H);
  bias_combine_kernel<<<(L * G + 255) / 256, 256, 0, stream>>>(bih, bhh, bias, L * G);

  for (int l = 0; l < L; ++l) {
    gemm_xg_kernel<<<dim3(G / BN, (S * B) / BM), 256, 0, stream>>>(
        l == 0 ? Xbf : seq0, Wihb + (size_t)l * G * H, bias + l * G, xg, l);
    init_state_kernel<<<(2 * B * H + 255) / 256, 256, 0, stream>>>(hbuf, cnt,
                                                                   2 * B * H);
    lstm_recurrent_kernel<<<NWG, 256, SMEM_BYTES, stream>>>(
        xg, Whhb + (size_t)l * G * H, hbuf,
        l == 0 ? seq0 : nullptr,
        l == 1 ? out : nullptr,
        cnt);
  }
}
